// RelativeAttention_28363964023160
// MI455X (gfx1250) — matrix operand order repaired and checked by reading, not compiled, not run on hardware
//
#include <hip/hip_runtime.h>

typedef _Float16 half_t;
typedef __attribute__((ext_vector_type(16))) _Float16 v16h;
typedef __attribute__((ext_vector_type(8)))  _Float16 v8h;
typedef __attribute__((ext_vector_type(8)))  float    v8f;
typedef __attribute__((vector_size(16)))     int      v4i;

#define D_MODEL 1024
#define D_HEAD  64
#define N_HEADS 16
#define SEQ     2048
#define BATCH   2
#define NTOK    (BATCH * SEQ)   // 4096
#define LOG2E   1.44269504088896f

#if __has_builtin(__builtin_amdgcn_global_load_async_to_lds_b128)
#define HAVE_ASYNC_LDS 1
#else
#define HAVE_ASYNC_LDS 0
#endif

// ---------------------------------------------------------------------------
// helpers
// ---------------------------------------------------------------------------

__device__ __forceinline__ void wait_async0() {
#if __has_builtin(__builtin_amdgcn_s_wait_asynccnt)
    __builtin_amdgcn_s_wait_asynccnt(0);
#else
    asm volatile("s_wait_asynccnt 0" ::: "memory");
#endif
}

#if HAVE_ASYNC_LDS
__device__ __forceinline__ void async_cp16(const half_t* gsrc, half_t* ldst) {
    __builtin_amdgcn_global_load_async_to_lds_b128(
        (__attribute__((address_space(1))) v4i*)gsrc,
        (__attribute__((address_space(3))) v4i*)ldst, 0, 0);
}
#endif

// Load a 16x32 f16 A-fragment: lane row m = lane&15,
// halves p=0..15 -> k = hi*16 + p (matches contiguous B fragments)
__device__ __forceinline__ v16h load_a_frag(const half_t* rowk0, int hi) {
    v16h a = *(const v16h*)(rowk0 + hi * 16);
    return a;
}

__device__ __forceinline__ v8f wmma_f16(v16h a, v16h b, v8f c) {
    return __builtin_amdgcn_wmma_f32_16x16x32_f16(false, a, false, b, (short)0, c,
                                                  false, false);
}

// reduce across the 16 lanes of a half-wave (xor masks stay within group of 16)
__device__ __forceinline__ float rowmax16(float x) {
    x = fmaxf(x, __int_as_float(__builtin_amdgcn_ds_swizzle(__float_as_int(x), 0x041F)));
    x = fmaxf(x, __int_as_float(__builtin_amdgcn_ds_swizzle(__float_as_int(x), 0x081F)));
    x = fmaxf(x, __int_as_float(__builtin_amdgcn_ds_swizzle(__float_as_int(x), 0x101F)));
    x = fmaxf(x, __int_as_float(__builtin_amdgcn_ds_swizzle(__float_as_int(x), 0x201F)));
    return x;
}
__device__ __forceinline__ float rowsum16(float x) {
    x += __int_as_float(__builtin_amdgcn_ds_swizzle(__float_as_int(x), 0x041F));
    x += __int_as_float(__builtin_amdgcn_ds_swizzle(__float_as_int(x), 0x081F));
    x += __int_as_float(__builtin_amdgcn_ds_swizzle(__float_as_int(x), 0x101F));
    x += __int_as_float(__builtin_amdgcn_ds_swizzle(__float_as_int(x), 0x201F));
    return x;
}

// ---------------------------------------------------------------------------
// f32 -> f16 conversion
// ---------------------------------------------------------------------------
__global__ void cvt_f32_f16(const float* __restrict__ src, half_t* __restrict__ dst,
                            int n) {
    int i = blockIdx.x * blockDim.x + threadIdx.x;
    int stride = gridDim.x * blockDim.x;
    for (; i < n; i += stride) dst[i] = (half_t)src[i];
}

// ---------------------------------------------------------------------------
// Shared GEMM core: each wave accumulates a 16x64 tile of  X @ W^T  over K=1024.
// When the CDNA5 async-to-LDS builtins exist, the 64x32 B tile is staged in a
// double-buffered LDS tile via GLOBAL_LOAD_ASYNC_TO_LDS_B128 (ASYNCcnt) shared
// by all 8 waves; otherwise each wave streams B straight from global (L0/L2).
// bs must point to 2 * 64*32 halves of LDS.
// ---------------------------------------------------------------------------
__device__ __forceinline__ void gemm_tile_16x64(const half_t* __restrict__ arow,
                                                const half_t* __restrict__ W, int n0,
                                                half_t* bs, v8f acc[4]) {
    const int tid  = threadIdx.x;
    const int lane = tid & 31;
    const int hi   = lane >> 4;
    const int ln   = lane & 15;
#if HAVE_ASYNC_LDS
    const int nl = tid >> 2;  // 0..63 : B row within tile
    const int qq = tid & 3;   // 16-byte quarter of the 64B row
    const half_t* gsrc0 = W + (size_t)(n0 + nl) * D_MODEL + qq * 8;
    half_t*       ldst0 = bs + nl * 32 + qq * 8;
    async_cp16(gsrc0, ldst0);  // prologue: k0 = 0 -> buffer 0
    wait_async0();
    __syncthreads();
    for (int k0 = 0; k0 < D_MODEL; k0 += 32) {
        const int cur = (k0 >> 5) & 1;
        if (k0 + 32 < D_MODEL)
            async_cp16(gsrc0 + k0 + 32, bs + (cur ^ 1) * (64 * 32) + nl * 32 + qq * 8);
        v16h a = load_a_frag(arow + k0, hi);
        const half_t* bbase = bs + cur * (64 * 32);
#pragma unroll
        for (int t = 0; t < 4; ++t) {
            v16h b = *(const v16h*)(bbase + (t * 16 + ln) * 32 + hi * 16);
            acc[t] = wmma_f16(a, b, acc[t]);
        }
        if (k0 + 32 < D_MODEL) wait_async0();
        __syncthreads();
    }
#else
    (void)bs;
    for (int k0 = 0; k0 < D_MODEL; k0 += 32) {
        v16h a = load_a_frag(arow + k0, hi);
#pragma unroll
        for (int t = 0; t < 4; ++t) {
            const half_t* brow = W + (size_t)(n0 + t * 16 + ln) * D_MODEL + k0 + hi * 16;
            v16h b = *(const v16h*)brow;
            acc[t] = wmma_f16(a, b, acc[t]);
        }
    }
#endif
}

// ---------------------------------------------------------------------------
// QKV projection: C = Xh @ W^T + b.   blockIdx.z selects Q/K/V.
// Q,K stored [b][h][s][64] f16;  V stored transposed [b][h][64][s] f16.
// block = 256 thr (8 waves); wave tile 16x64; block tile 128 rows x 64 cols.
// blockIdx.y == head.
// ---------------------------------------------------------------------------
__global__ __launch_bounds__(256) void qkv_gemm(
    const half_t* __restrict__ xh, const half_t* __restrict__ wq,
    const half_t* __restrict__ wk, const half_t* __restrict__ wv,
    const float* __restrict__ bq, const float* __restrict__ bk,
    const float* __restrict__ bv, half_t* __restrict__ Qo,
    half_t* __restrict__ Ko, half_t* __restrict__ Vt) {
    __shared__ half_t bsmem[2][64 * 32];
    const int lane = threadIdx.x & 31;
    const int wid  = threadIdx.x >> 5;
    const int hi   = lane >> 4;
    const int ln   = lane & 15;
    const int m0   = blockIdx.x * 128 + wid * 16;
    const int h    = blockIdx.y;
    const int n0   = h * 64;
    const int proj = blockIdx.z;

    const half_t* W    = proj == 0 ? wq : (proj == 1 ? wk : wv);
    const float*  bias = proj == 0 ? bq : (proj == 1 ? bk : bv);

    v8f acc[4] = {};
    gemm_tile_16x64(xh + (size_t)(m0 + ln) * D_MODEL, W, n0, &bsmem[0][0], acc);

    if (proj < 2) {
        half_t* O = proj == 0 ? Qo : Ko;
#pragma unroll
        for (int t = 0; t < 4; ++t) {
            float bb = bias[n0 + t * 16 + ln];
            int   d  = t * 16 + ln;
#pragma unroll
            for (int v = 0; v < 8; ++v) {
                int tok = m0 + v + 8 * hi;
                int bi = tok >> 11, s = tok & (SEQ - 1);
                O[((size_t)(bi * N_HEADS + h) * SEQ + s) * D_HEAD + d] =
                    (half_t)(acc[t][v] + bb);
            }
        }
    } else {
        int tok0 = m0 + 8 * hi;
        int bi = tok0 >> 11, s0 = tok0 & (SEQ - 1);
#pragma unroll
        for (int t = 0; t < 4; ++t) {
            float bb = bias[n0 + t * 16 + ln];
            int   d  = t * 16 + ln;
            v8h pk;
#pragma unroll
            for (int v = 0; v < 8; ++v) pk[v] = (half_t)(acc[t][v] + bb);
            *(v8h*)(Vt + ((size_t)(bi * N_HEADS + h) * D_HEAD + d) * SEQ + s0) = pk;
        }
    }
}

// ---------------------------------------------------------------------------
// Flash attention with relative-position term.
// One wave per (b, h, 16-query tile); key blocks of 32.
// rel[i,j] = q_i . Er[S-1-(i-j)] via 16x48 band GEMM + bpermute diagonal gather.
// Writes ctx [b*S+s][h*64+d] f16.
// ---------------------------------------------------------------------------
__global__ __launch_bounds__(256) void rel_attention(
    const half_t* __restrict__ Q, const half_t* __restrict__ K,
    const half_t* __restrict__ Vt, const half_t* __restrict__ Erh,
    half_t* __restrict__ ctx) {
    __shared__ half_t lds_p[8][16 * 32];  // per-wave P staging (C-layout -> A-layout)

    const int lane = threadIdx.x & 31;
    const int wid  = threadIdx.x >> 5;
    const int hi   = lane >> 4;
    const int ln   = lane & 15;

    const int tile = blockIdx.x * 8 + wid;  // 4096 tiles total
    const int b  = tile >> 11;
    const int h  = (tile >> 7) & 15;
    const int qt = tile & 127;
    const int i0 = qt * 16;

    const half_t* qp = Q + (size_t)(b * N_HEADS + h) * SEQ * D_HEAD;
    const half_t* kp = K + (size_t)(b * N_HEADS + h) * SEQ * D_HEAD;
    const half_t* vp = Vt + (size_t)(b * N_HEADS + h) * D_HEAD * SEQ;

    // Q A-fragments (row m = i0+ln), k = 0..31 and 32..63
    const half_t* qrow = qp + (size_t)(i0 + ln) * D_HEAD;
    v16h qa0 = load_a_frag(qrow, hi);
    v16h qa1 = load_a_frag(qrow + 32, hi);

    v8f   o[4] = {};
    float rmax[8], rsum[8];
#pragma unroll
    for (int v = 0; v < 8; ++v) { rmax[v] = -3.0e38f; rsum[v] = 0.0f; }

    half_t* pb = lds_p[wid];
    const int nblk = (i0 + 47) >> 5;

    for (int kb = 0; kb < nblk; ++kb) {
        const int jb = kb * 32;
        if (kb + 1 < nblk)  // global_prefetch_b8 for next K block
            __builtin_prefetch(kp + (size_t)(jb + 32 + ln) * D_HEAD, 0, 0);

        // ---- scores: QK^T for two 16-key subtiles
        v8f sc[2] = {};
#pragma unroll
        for (int s = 0; s < 2; ++s) {
            const half_t* krow = kp + (size_t)(jb + s * 16 + ln) * D_HEAD + hi * 16;
            v16h b0 = *(const v16h*)krow;
            v16h b1 = *(const v16h*)(krow + 32);
            sc[s] = wmma_f16(qa0, b0, sc[s]);
            sc[s] = wmma_f16(qa1, b1, sc[s]);
        }

        // ---- relative band: Rel[m, c] = Q[m] . Er[base-15+c], c in [0,48)
        const int base = SEQ - 1 - i0 + jb;
        v8f rl[3] = {};
#pragma unroll
        for (int t = 0; t < 3; ++t) {
            int r = base - 15 + t * 16 + ln;
            r = r < 0 ? 0 : (r > SEQ - 1 ? SEQ - 1 : r);  // clamped rows are masked
            const half_t* erow = Erh + (size_t)r * D_HEAD + hi * 16;
            v16h e0 = *(const v16h*)erow;
            v16h e1 = *(const v16h*)(erow + 32);
            rl[t] = wmma_f16(qa0, e0, rl[t]);
            rl[t] = wmma_f16(qa1, e1, rl[t]);
        }

        const bool needmask = (jb + 31 > i0);

        // ---- gather diagonals, scale, mask
        // subtile 0: c in [0,30] -> bands {0,1}; subtile 1: c in [16,46] -> {1,2}
#pragma unroll
        for (int s = 0; s < 2; ++s) {
#pragma unroll
            for (int v = 0; v < 8; ++v) {
                int m = v + 8 * hi;
                int c = ln + 16 * s - m + 15;
                int srcaddr = (((c & 15) | (lane & 16)) << 2);
                float glo = __int_as_float(
                    __builtin_amdgcn_ds_bpermute(srcaddr, __float_as_int(rl[s][v])));
                float ghi = __int_as_float(
                    __builtin_amdgcn_ds_bpermute(srcaddr, __float_as_int(rl[s + 1][v])));
                float rel = (c < 16 * (s + 1)) ? glo : ghi;
                float scv = (sc[s][v] + rel) * 0.125f;  // / sqrt(64)
                if (needmask && (jb + 16 * s + ln > i0 + m)) scv = -1.0e30f;
                sc[s][v] = scv;
            }
        }

        // ---- online softmax per row + write P (f16) to LDS in [m][k] order
#pragma unroll
        for (int v = 0; v < 8; ++v) {
            float bm    = rowmax16(fmaxf(sc[0][v], sc[1][v]));
            float mnew  = fmaxf(rmax[v], bm);
            float alpha = exp2f((rmax[v] - mnew) * LOG2E);
            float p0 = exp2f((sc[0][v] - mnew) * LOG2E);
            float p1 = exp2f((sc[1][v] - mnew) * LOG2E);
            rsum[v] = rsum[v] * alpha + rowsum16(p0 + p1);
            rmax[v] = mnew;
            int m = v + 8 * hi;
            pb[m * 32 + ln]      = (half_t)p0;
            pb[m * 32 + 16 + ln] = (half_t)p1;
#pragma unroll
            for (int dt = 0; dt < 4; ++dt) o[dt][v] *= alpha;
        }

        // ---- O += P @ V  (A = P from LDS, B = Vt rows, contiguous in keys)
        v16h pa = load_a_frag(pb + ln * 32, hi);
#pragma unroll
        for (int dt = 0; dt < 4; ++dt) {
            const half_t* vrow = vp + (size_t)(dt * 16 + ln) * SEQ + jb + hi * 16;
            v16h vb = *(const v16h*)vrow;
            o[dt] = wmma_f16(pa, vb, o[dt]);
        }
    }

    // ---- normalize and write ctx [tok][h*64+d]
#pragma unroll
    for (int dt = 0; dt < 4; ++dt) {
        int d = dt * 16 + ln;
#pragma unroll
        for (int v = 0; v < 8; ++v) {
            int m = v + 8 * hi;
            float val = o[dt][v] / rsum[v];
            ctx[((size_t)(b * SEQ) + i0 + m) * D_MODEL + h * D_HEAD + d] = (half_t)val;
        }
    }
}

// ---------------------------------------------------------------------------
// Output projection: out = ctx @ Wo^T + bo  (f32 output)
// ---------------------------------------------------------------------------
__global__ __launch_bounds__(256) void out_proj(const half_t* __restrict__ ctx,
                                                const half_t* __restrict__ wo,
                                                const float* __restrict__ bo,
                                                float* __restrict__ out) {
    __shared__ half_t bsmem[2][64 * 32];
    const int lane = threadIdx.x & 31;
    const int wid  = threadIdx.x >> 5;
    const int hi   = lane >> 4;
    const int ln   = lane & 15;
    const int m0   = blockIdx.x * 128 + wid * 16;
    const int n0   = blockIdx.y * 64;

    v8f acc[4] = {};
    gemm_tile_16x64(ctx + (size_t)(m0 + ln) * D_MODEL, wo, n0, &bsmem[0][0], acc);

#pragma unroll
    for (int t = 0; t < 4; ++t) {
        int   n  = n0 + t * 16 + ln;
        float bb = bo[n];
#pragma unroll
        for (int v = 0; v < 8; ++v) {
            int tok = m0 + v + 8 * hi;
            out[(size_t)tok * D_MODEL + n] = acc[t][v] + bb;
        }
    }
}

// ---------------------------------------------------------------------------
extern "C" void kernel_launch(void* const* d_in, const int* in_sizes, int n_in,
                              void* d_out, int out_size, void* d_ws, size_t ws_size,
                              hipStream_t stream) {
    (void)in_sizes; (void)n_in; (void)out_size; (void)ws_size;
    const float* x  = (const float*)d_in[0];
    // d_in[1] = mask (causal, implemented analytically)
    const float* Wq = (const float*)d_in[2];
    const float* bq = (const float*)d_in[3];
    const float* Wk = (const float*)d_in[4];
    const float* bk = (const float*)d_in[5];
    const float* Wv = (const float*)d_in[6];
    const float* bv = (const float*)d_in[7];
    const float* Er = (const float*)d_in[8];
    const float* Wo = (const float*)d_in[9];
    const float* bo = (const float*)d_in[10];

    char*  ws  = (char*)d_ws;
    size_t off = 0;
    auto carve = [&](size_t bytes) -> void* {
        void* p = ws + off;
        off += (bytes + 255) & ~(size_t)255;
        return p;
    };
    half_t* xh   = (half_t*)carve((size_t)NTOK * D_MODEL * 2);
    half_t* qbuf = (half_t*)carve((size_t)NTOK * D_MODEL * 2);
    half_t* kbuf = (half_t*)carve((size_t)NTOK * D_MODEL * 2);
    half_t* vtbf = (half_t*)carve((size_t)NTOK * D_MODEL * 2);
    half_t* cbuf = (half_t*)carve((size_t)NTOK * D_MODEL * 2);
    half_t* wqh  = (half_t*)carve((size_t)D_MODEL * D_MODEL * 2);
    half_t* wkh  = (half_t*)carve((size_t)D_MODEL * D_MODEL * 2);
    half_t* wvh  = (half_t*)carve((size_t)D_MODEL * D_MODEL * 2);
    half_t* woh  = (half_t*)carve((size_t)D_MODEL * D_MODEL * 2);
    half_t* erh  = (half_t*)carve((size_t)SEQ * D_HEAD * 2);

    cvt_f32_f16<<<4096, 256, 0, stream>>>(x, xh, NTOK * D_MODEL);
    cvt_f32_f16<<<2048, 256, 0, stream>>>(Wq, wqh, D_MODEL * D_MODEL);
    cvt_f32_f16<<<2048, 256, 0, stream>>>(Wk, wkh, D_MODEL * D_MODEL);
    cvt_f32_f16<<<2048, 256, 0, stream>>>(Wv, wvh, D_MODEL * D_MODEL);
    cvt_f32_f16<<<2048, 256, 0, stream>>>(Wo, woh, D_MODEL * D_MODEL);
    cvt_f32_f16<<<512, 256, 0, stream>>>(Er, erh, SEQ * D_HEAD);

    dim3 gq(NTOK / 128, N_HEADS, 3);
    qkv_gemm<<<gq, 256, 0, stream>>>(xh, wqh, wkh, wvh, bq, bk, bv, qbuf, kbuf, vtbf);

    rel_attention<<<(BATCH * N_HEADS * (SEQ / 16)) / 8, 256, 0, stream>>>(
        qbuf, kbuf, vtbf, erh, cbuf);

    dim3 go(NTOK / 128, D_MODEL / 64);
    out_proj<<<go, 256, 0, stream>>>(cbuf, woh, bo, (float*)d_out);
}
